// HVGGBlock_44976897523753
// MI455X (gfx1250) — matrix operand order repaired and checked by reading, not compiled, not run on hardware
//
#include <hip/hip_runtime.h>
#include <hip/hip_bf16.h>

typedef __attribute__((ext_vector_type(16))) __bf16 v16bf;
typedef __attribute__((ext_vector_type(8)))  float  v8f;
typedef __attribute__((ext_vector_type(4)))  unsigned v4u;
typedef __attribute__((ext_vector_type(8)))  int    v8i;
typedef __attribute__((ext_vector_type(4)))  int    v4i;

#define K_NB   8
#define K_C    32
#define K_H    256
#define K_W    256
#define K_HW   (K_H * K_W)
#define K_NPIX (K_NB * K_HW)     // 524288 pixels
#define EPSI   1e-5f
#define MAXN   0.9999f           // (1 - 1e-4) / sqrt(c), c = 1

// ---------- helpers ----------
__device__ __forceinline__ unsigned short f2bf(float f) {
    unsigned u = __float_as_uint(f);
    unsigned r = (u + 0x7FFFu + ((u >> 16) & 1u)) >> 16;   // RNE
    return (unsigned short)r;
}
__device__ __forceinline__ float artanh_c(float x) {
    x = fminf(fmaxf(x, -0.99999f), 0.99999f);
    return 0.5f * (log1pf(x) - log1pf(-x));
}
// gfx1250 async global->LDS b128 copy (per-lane, ASYNCcnt-tracked)
__device__ __forceinline__ void async_g2l_b128(unsigned lds_off, unsigned long long gaddr) {
    asm volatile("global_load_async_to_lds_b128 %0, %1, off"
                 :: "v"(lds_off), "v"(gaddr) : "memory");
}
__device__ __forceinline__ void wait_asynccnt0() {
    asm volatile("s_wait_asynccnt 0x0" ::: "memory");
}
// TDM: DMA nelem8B x 8 bytes global -> LDS, issued by one wave (EXEC-ignored DMA).
// D# per CDNA5 ISA ch.8: group0 {count=1, lds_addr, global_addr[56:0], type=2},
// group1 {data_size=8B, dim0=tile0=stride0=nelem, dim1=tile1=1}, groups 2/3 zero (<=2D).
__device__ __forceinline__ void tdm_load_to_lds(unsigned lds_addr, unsigned long long gaddr,
                                                unsigned nelem8B) {
    v4u g0;
    g0[0] = 1u;                                   // count=1, user descriptor
    g0[1] = lds_addr;                             // LDS byte address
    g0[2] = (unsigned)(gaddr & 0xffffffffu);      // global addr [31:0]
    g0[3] = (unsigned)(gaddr >> 32) | (2u << 30); // global addr [56:32] | type=2
    v8i g1;
    g1[0] = (3 << 16);                            // workgroup_mask=0, data_size=3 (8 bytes)
    g1[1] = (int)((nelem8B & 0xffffu) << 16);     // tensor_dim0 low16
    g1[2] = (int)((nelem8B >> 16) | (1u << 16));  // tensor_dim0 hi16 | tensor_dim1=1 lo16
    g1[3] = (int)((nelem8B & 0xffffu) << 16);     // tensor_dim1 hi=0 | tile_dim0
    g1[4] = 1;                                    // tile_dim1=1, tile_dim2=0
    g1[5] = (int)nelem8B;                         // tensor_dim0_stride low32
    g1[6] = 0;                                    // stride hi | tensor_dim1_stride lo
    g1[7] = 0;
    v4i gz4 = {0, 0, 0, 0};
    v8i gz8 = {0, 0, 0, 0, 0, 0, 0, 0};
    __builtin_amdgcn_tensor_load_to_lds(g0, g1, gz4, gz4, gz8, 0);   // 6-arg toolchain form
}

// ---------- weight prep: OIHW f32 -> WMMA-B-fragment bf16 layout ----------
// wb[((t*2+nt)*32 + lane)*16 + e] = bf16(w[o][ci][ky][kx])
//   o = (lane&15) + 16*nt, ci = 16*(lane>>4) + e, t = ky*3+kx
__global__ void prep_w(const float* __restrict__ w, unsigned short* __restrict__ wb) {
    int j = blockIdx.x * blockDim.x + threadIdx.x;
    if (j >= 9 * 2 * 32 * 16) return;
    int e    = j & 15;
    int lane = (j >> 4) & 31;
    int nt   = (j >> 9) & 1;
    int t    = j >> 10;
    int o  = (lane & 15) + 16 * nt;
    int ci = 16 * (lane >> 4) + e;
    int ky = t / 3, kx = t % 3;
    wb[j] = f2bf(w[((o * 32 + ci) * 3 + ky) * 3 + kx]);
}

// ---------- K1: u = logmap0(x), NCHW f32 -> NHWC bf16 ----------
__global__ void logmap_in(const float* __restrict__ x, unsigned short* __restrict__ U) {
    int idx = blockIdx.x * blockDim.x + threadIdx.x;   // pixel id
    int n = idx >> 16, rem = idx & 65535;
    const float* xp = x + (size_t)n * K_C * K_HW + rem;
    float v[32]; float ss = 0.f;
    #pragma unroll
    for (int c = 0; c < 32; c++) { float t = xp[(size_t)c * K_HW]; v[c] = t; ss += t * t; }
    float nn = fmaxf(sqrtf(ss), EPSI);
    float s  = artanh_c(nn) / nn;
    unsigned short* up = U + (size_t)idx * 32;
    #pragma unroll
    for (int c = 0; c < 32; c += 8) {
        uint4 q;
        q.x = (unsigned)f2bf(s*v[c+0]) | ((unsigned)f2bf(s*v[c+1]) << 16);
        q.y = (unsigned)f2bf(s*v[c+2]) | ((unsigned)f2bf(s*v[c+3]) << 16);
        q.z = (unsigned)f2bf(s*v[c+4]) | ((unsigned)f2bf(s*v[c+5]) << 16);
        q.w = (unsigned)f2bf(s*v[c+6]) | ((unsigned)f2bf(s*v[c+7]) << 16);
        *(uint4*)(up + c) = q;
    }
}

// ---------- K2: 3x3 conv via implicit GEMM, bf16 WMMA, f32 accum ----------
// Block = 128 threads (4 waves), processes 4 output rows x 64 cols x 32 Cout.
// Weights: one TDM descriptor (wave 0) -> 18432 B LDS.
// Tile: 6 rows x 66 cols x 32ch bf16, per-lane async-to-LDS with zero halo.
// grid = N * (H/4) * (W/64) = 2048 blocks.
__global__ __launch_bounds__(128) void hconv_wmma(
        const unsigned short* __restrict__ U,   // NHWC bf16 tangent input
        const unsigned short* __restrict__ Wb,  // prepped B fragments
        const float* __restrict__ bias,
        float* __restrict__ Y) {                // NHWC f32 out (v + bias)
    __shared__ unsigned short lw[9 * 2 * 32 * 16];   // 18432 B
    __shared__ unsigned short lu[6 * 66 * 32];       // 25344 B (6 rows, 66 cols halo)
    int tid  = threadIdx.x;
    int nb   = blockIdx.x;
    int wseg = nb & 3;
    int h0   = ((nb >> 2) & 63) * 4;
    int n    = nb >> 8;
    int w0   = wseg * 64;

    unsigned lwbase = (unsigned)(uintptr_t)(&lw[0]);
    unsigned lubase = (unsigned)(uintptr_t)(&lu[0]);

    // stage weights: single TDM DMA (2304 x 8B), issued by wave 0 only
    if (tid < 32) {
        tdm_load_to_lds(lwbase, (unsigned long long)(uintptr_t)Wb, 2304u);
    }
    // stage input tile with zero halo: 396 slots x 4 x b128, per-lane async
    for (int i = tid; i < 396 * 4; i += 128) {
        int slot = i >> 2, part = i & 3;
        int r = slot / 66, cc = slot % 66;
        int hh = h0 - 1 + r, ww = w0 - 1 + cc;
        unsigned loff = lubase + (unsigned)(slot * 64 + part * 16);
        if ((unsigned)hh < 256u && (unsigned)ww < 256u) {
            size_t pix = ((size_t)n * 256 + hh) * 256 + ww;
            async_g2l_b128(loff,
                (unsigned long long)(uintptr_t)(U + pix * 32 + part * 8));
        } else {
            uint4 z; z.x = z.y = z.z = z.w = 0u;
            *(uint4*)(lu + slot * 32 + part * 8) = z;   // zero pad via ds_store
        }
    }
    wait_asynccnt0();                       // drain this wave's async copies
    if (tid < 32) {
        __builtin_amdgcn_s_wait_tensorcnt((short)0);   // wave 0: drain TDM before barrier
    }
    __syncthreads();                        // publishes LDS to all waves

    int wv = tid >> 5, lane = tid & 31;
    int half = lane >> 4, l15 = lane & 15;
    int wloc = wv * 16 + l15;                     // A-row pixel column in segment
    float bA = bias[l15], bB = bias[l15 + 16];

    union AF { v16bf v; uint4 q[2]; };
    for (int r = 0; r < 4; r++) {                 // 4 output rows per block
        v8f acc0 = {}; v8f acc1 = {};
        #pragma unroll
        for (int t = 0; t < 9; t++) {
            int dy = t / 3, dx = t % 3;
            // A fragment (16x32 bf16): lane half selects K sub-range, contiguous k = 16*half + e (as in B prep)
            const unsigned short* ap = lu + ((r + dy) * 66 + (wloc + dx)) * 32;
            AF a;
            a.q[0] = *(const uint4*)(ap + half * 16);       // ch 16h..16h+7
            a.q[1] = *(const uint4*)(ap + half * 16 + 8);   // ch 16h+8..16h+15
            // B fragments (32x16 bf16), two N-tiles
            const unsigned short* bp0 = lw + ((t * 2 + 0) * 32 + lane) * 16;
            const unsigned short* bp1 = lw + ((t * 2 + 1) * 32 + lane) * 16;
            AF b0, b1;
            b0.q[0] = *(const uint4*)(bp0);  b0.q[1] = *(const uint4*)(bp0 + 8);
            b1.q[0] = *(const uint4*)(bp1);  b1.q[1] = *(const uint4*)(bp1 + 8);
            acc0 = __builtin_amdgcn_wmma_f32_16x16x32_bf16(false, a.v, false, b0.v,
                                                           (short)0, acc0, false, false);
            acc1 = __builtin_amdgcn_wmma_f32_16x16x32_bf16(false, a.v, false, b1.v,
                                                           (short)0, acc1, false, false);
        }
        size_t rowbase = ((size_t)n * 256 + (h0 + r)) * 256 + w0;
        #pragma unroll
        for (int m = 0; m < 8; m++) {             // D layout: M = m + 8*half, N = l15
            size_t pix = rowbase + wv * 16 + half * 8 + m;
            Y[pix * 32 + l15]      = acc0[m] + bA;
            Y[pix * 32 + 16 + l15] = acc1[m] + bB;
        }
    }
}

// ---------- K3: u = logmap0(project(expmap0(v))) in place + BN stats ----------
__global__ void tangent_stats(float* __restrict__ Y, float* __restrict__ stats) {
    __shared__ float ls[64];
    int tid = threadIdx.x;
    if (tid < 64) ls[tid] = 0.f;
    __syncthreads();
    int idx = blockIdx.x * blockDim.x + tid;
    float* yp = Y + (size_t)idx * 32;
    float v[32]; float ss = 0.f;
    #pragma unroll
    for (int j = 0; j < 8; j++) {
        float4 q = *(const float4*)(yp + 4 * j);
        v[4*j+0] = q.x; v[4*j+1] = q.y; v[4*j+2] = q.z; v[4*j+3] = q.w;
    }
    #pragma unroll
    for (int c = 0; c < 32; c++) ss += v[c] * v[c];
    float n1 = fmaxf(sqrtf(ss), EPSI);
    float s  = artanh_c(fminf(tanhf(n1), MAXN)) / n1;  // exp->project->log composite
    #pragma unroll
    for (int c = 0; c < 32; c++) v[c] *= s;
    #pragma unroll
    for (int j = 0; j < 8; j++) {
        float4 q; q.x = v[4*j+0]; q.y = v[4*j+1]; q.z = v[4*j+2]; q.w = v[4*j+3];
        *(float4*)(yp + 4 * j) = q;
    }
    #pragma unroll
    for (int c = 0; c < 32; c++) {
        atomicAdd(&ls[c], v[c]);
        atomicAdd(&ls[32 + c], v[c] * v[c]);
    }
    __syncthreads();
    if (tid < 64) atomicAdd(&stats[tid], ls[tid]);
}

__global__ void zero_stats(float* stats) {
    if (threadIdx.x < 64) stats[threadIdx.x] = 0.f;
}

__global__ void finalize_stats(float* stats) {
    int c = threadIdx.x;
    if (c < 32) {
        float cnt = (float)K_NPIX;
        float m   = stats[c] / cnt;
        float var = stats[32 + c] / cnt - m * m;
        stats[64 + c] = m;
        stats[96 + c] = rsqrtf(var + 1e-5f);
    }
}

// ---------- K5: BN affine + (exp/proj/log) + relu + (exp/proj/log) -> bf16 NHWC ----------
__global__ void bn_relu_tangent(const float* __restrict__ Yu, const float* __restrict__ stats,
                                const float* __restrict__ g, const float* __restrict__ beta,
                                unsigned short* __restrict__ U) {
    int idx = blockIdx.x * blockDim.x + threadIdx.x;
    const float* up = Yu + (size_t)idx * 32;
    float v[32]; float ss = 0.f;
    #pragma unroll
    for (int c = 0; c < 32; c++) {
        float t = (up[c] - stats[64 + c]) * stats[96 + c] * g[c] + beta[c];
        v[c] = t; ss += t * t;
    }
    float n1 = fmaxf(sqrtf(ss), EPSI);
    float a1 = artanh_c(fminf(tanhf(n1), MAXN)) / n1;   // hbn out -> hrelu tangent
    float ss2 = 0.f;
    #pragma unroll
    for (int c = 0; c < 32; c++) { float t = fmaxf(a1 * v[c], 0.f); v[c] = t; ss2 += t * t; }
    float n2 = fmaxf(sqrtf(ss2), EPSI);
    float a2 = artanh_c(fminf(tanhf(n2), MAXN)) / n2;   // hrelu out -> next logmap0
    unsigned short* op = U + (size_t)idx * 32;
    #pragma unroll
    for (int c = 0; c < 32; c += 8) {
        uint4 q;
        q.x = (unsigned)f2bf(a2*v[c+0]) | ((unsigned)f2bf(a2*v[c+1]) << 16);
        q.y = (unsigned)f2bf(a2*v[c+2]) | ((unsigned)f2bf(a2*v[c+3]) << 16);
        q.z = (unsigned)f2bf(a2*v[c+4]) | ((unsigned)f2bf(a2*v[c+5]) << 16);
        q.w = (unsigned)f2bf(a2*v[c+6]) | ((unsigned)f2bf(a2*v[c+7]) << 16);
        *(uint4*)(op + c) = q;
    }
}

// ---------- K9: BN affine + (exp/proj/log) + relu + final expmap+project -> NCHW f32 ----------
__global__ void bn_relu_out(const float* __restrict__ Yu, const float* __restrict__ stats,
                            const float* __restrict__ g, const float* __restrict__ beta,
                            float* __restrict__ out) {
    int idx = blockIdx.x * blockDim.x + threadIdx.x;
    int n = idx >> 16, rem = idx & 65535;
    const float* up = Yu + (size_t)idx * 32;
    float v[32]; float ss = 0.f;
    #pragma unroll
    for (int c = 0; c < 32; c++) {
        float t = (up[c] - stats[64 + c]) * stats[96 + c] * g[c] + beta[c];
        v[c] = t; ss += t * t;
    }
    float n1 = fmaxf(sqrtf(ss), EPSI);
    float a1 = artanh_c(fminf(tanhf(n1), MAXN)) / n1;
    float ss2 = 0.f;
    #pragma unroll
    for (int c = 0; c < 32; c++) { float t = fmaxf(a1 * v[c], 0.f); v[c] = t; ss2 += t * t; }
    float n2 = fmaxf(sqrtf(ss2), EPSI);
    float a2 = fminf(tanhf(n2), MAXN) / n2;             // final expmap0 + project
    float* op = out + (size_t)n * K_C * K_HW + rem;
    #pragma unroll
    for (int c = 0; c < 32; c++) op[(size_t)c * K_HW] = a2 * v[c];
}

// ---------- launch ----------
extern "C" void kernel_launch(void* const* d_in, const int* in_sizes, int n_in,
                              void* d_out, int out_size, void* d_ws, size_t ws_size,
                              hipStream_t stream) {
    const float* x   = (const float*)d_in[0];
    const float* w1  = (const float*)d_in[1];
    const float* b1  = (const float*)d_in[2];
    const float* g1  = (const float*)d_in[3];
    const float* be1 = (const float*)d_in[4];
    const float* w2  = (const float*)d_in[5];
    const float* b2  = (const float*)d_in[6];
    const float* g2  = (const float*)d_in[7];
    const float* be2 = (const float*)d_in[8];

    char* ws = (char*)d_ws;
    unsigned short* U   = (unsigned short*)(ws);                       // 32 MiB  bf16 NHWC
    float*          Y   = (float*)(ws + 33554432);                     // 64 MiB  f32  NHWC
    unsigned short* Wb1 = (unsigned short*)(ws + 100663296);           // 18 KiB
    unsigned short* Wb2 = (unsigned short*)(ws + 100663296 + 18432);   // 18 KiB
    float*          st  = (float*)(ws + 100663296 + 36864);            // 512 B

    const int EB = K_NPIX / 256;                    // 2048 elementwise blocks
    const int CB = K_NB * (K_H / 4) * (K_W / 64);   // 2048 conv blocks

    prep_w<<<36, 256, 0, stream>>>(w1, Wb1);
    prep_w<<<36, 256, 0, stream>>>(w2, Wb2);

    logmap_in<<<EB, 256, 0, stream>>>(x, U);

    hconv_wmma<<<CB, 128, 0, stream>>>(U, Wb1, b1, Y);
    zero_stats<<<1, 64, 0, stream>>>(st);
    tangent_stats<<<EB, 256, 0, stream>>>(Y, st);
    finalize_stats<<<1, 32, 0, stream>>>(st);
    bn_relu_tangent<<<EB, 256, 0, stream>>>(Y, st, g1, be1, U);

    hconv_wmma<<<CB, 128, 0, stream>>>(U, Wb2, b2, Y);
    zero_stats<<<1, 64, 0, stream>>>(st);
    tangent_stats<<<EB, 256, 0, stream>>>(Y, st);
    finalize_stats<<<1, 32, 0, stream>>>(st);
    bn_relu_out<<<EB, 256, 0, stream>>>(Y, st, g2, be2, (float*)d_out);
}
